// NNUEModel_4157528342874
// MI455X (gfx1250) — compile-verified, matrix operand order checked by reading, not run on hardware
//
#include <hip/hip_runtime.h>

// NNUE fused forward for MI455X (gfx1250, wave32).
// Phase 0: block stages w1 (512x32) / w2 (32x32) into LDS pre-swizzled into the
//          CDNA5 WMMA bf16 B-fragment layout.
// Phase 1: each wave gathers+sums 2x32 embedding rows for 16 batch rows
//          (lane owns an 8-float slice), packs bf16 A-fragments into LDS.
// Phase 2: layer1 via 32x v_wmma_f32_16x16x32_bf16, bias+ReLU -> LDS.
// Phase 3: layer2 via 2x WMMA (K=32 exactly), bias+ReLU -> LDS.
// Phase 4: layer3 dot-32 per row, store f32 output.

typedef __attribute__((ext_vector_type(16))) __bf16 v16bf;
typedef __attribute__((ext_vector_type(8)))  __bf16 v8bf;
typedef __attribute__((ext_vector_type(8)))  float  v8f;

#define WPB   4      // waves per block
#define RPW   16     // batch rows per wave
#define HDIM  256
#define NF    32     // active features per side
#define KC1   16     // 512 / 32 K-chunks for layer 1

__device__ __forceinline__ void add4(float4& a, const float4 b) {
  a.x += b.x; a.y += b.y; a.z += b.z; a.w += b.w;
}

__global__ __launch_bounds__(WPB * 32)
void nnue_fused_kernel(const int* __restrict__ widx,
                       const int* __restrict__ bidx,
                       const float* __restrict__ wtab,
                       const float* __restrict__ btab,
                       const float* __restrict__ w1,
                       const float* __restrict__ b1,
                       const float* __restrict__ w2,
                       const float* __restrict__ b2,
                       const float* __restrict__ w3,
                       const float* __restrict__ b3,
                       float* __restrict__ out)
{
  // LDS: weights in B-fragment layout + per-wave activation staging.
  __shared__ __align__(32) __bf16 w1f[KC1 * 2 * 32 * 16];   // 32 KB
  __shared__ __align__(32) __bf16 w2f[2 * 32 * 16];         //  2 KB
  __shared__ __align__(32) __bf16 xfrag[WPB][KC1 * 32 * 16];// 64 KB
  __shared__ __align__(16) float  hs1[WPB][16 * 32];        //  8 KB
  __shared__ __align__(16) float  hs2[WPB][16 * 32];        //  8 KB

  const int tid  = threadIdx.x;
  const int wave = tid >> 5;
  const int lane = tid & 31;

  // ---- Phase 0: stage w1/w2 as bf16 B-fragments ------------------------
  // B-fragment (16-bit, KxN=32x16): lane holds col n = (lane&15) + 16*nh,
  // K-half = (lane>>4)*8; elements 0..7 -> k = kc*32 + khalf + e,
  // elements 8..15 -> k = kc*32 + 16 + khalf + (e-8).
  for (int i = tid; i < KC1 * 2 * 32 * 16; i += blockDim.x) {
    int e  = i & 15;
    int ln = (i >> 4) & 31;
    int nh = (i >> 9) & 1;
    int kc = i >> 10;
    int n  = (ln & 15) + 16 * nh;
    int kh = (ln >> 4) * 8;
    int k  = kc * 32 + kh + ((e < 8) ? e : (16 + (e - 8)));
    w1f[i] = (__bf16)w1[k * 32 + n];
  }
  for (int i = tid; i < 2 * 32 * 16; i += blockDim.x) {
    int e  = i & 15;
    int ln = (i >> 4) & 31;
    int nh = i >> 9;
    int n  = (ln & 15) + 16 * nh;
    int kh = (ln >> 4) * 8;
    int k  = kh + ((e < 8) ? e : (16 + (e - 8)));
    w2f[i] = (__bf16)w2[k * 32 + n];
  }
  __syncthreads();

  const int row_base = blockIdx.x * (WPB * RPW) + wave * RPW;
  __bf16* xw = &xfrag[wave][0];

  // Per-lane constants mapping x-column slice [8*lane, 8*lane+8) into the
  // 16-bit A-fragment layout (A 16x32: lane' = M + 16*Khalf).
  const int kcw   = lane >> 2;             // white K-chunk (0..7); black = +8
  const int lhoff = (lane & 1) * 16;       // K-half -> lane' offset
  const int ebase = 8 * ((lane >> 1) & 1); // element base within fragment

  // ---- Phase 1: embedding-bag gather + sum (L2-resident tables) --------
  for (int r = 0; r < RPW; ++r) {
    const int row = row_base + r;
    const int wi  = widx[row * NF + lane];
    const int bi  = bidx[row * NF + lane];
    float4 aw0 = make_float4(0.f, 0.f, 0.f, 0.f);
    float4 aw1 = aw0, ab0 = aw0, ab1 = aw0;
    #pragma unroll 4
    for (int k = 0; k < NF; ++k) {
      const int iw = __shfl(wi, k, 32);
      const int ib = __shfl(bi, k, 32);
      const float4* wr = (const float4*)(wtab + (size_t)iw * HDIM);
      const float4* br = (const float4*)(btab + (size_t)ib * HDIM);
      float4 w0 = wr[lane * 2], w1v = wr[lane * 2 + 1];
      float4 c0 = br[lane * 2], c1v = br[lane * 2 + 1];
      add4(aw0, w0); add4(aw1, w1v);
      add4(ab0, c0); add4(ab1, c1v);
    }
    // Pack the 8 white / 8 black sums to bf16 and scatter into A-fragments.
    v8bf pw, pb;
    pw[0] = (__bf16)aw0.x; pw[1] = (__bf16)aw0.y; pw[2] = (__bf16)aw0.z; pw[3] = (__bf16)aw0.w;
    pw[4] = (__bf16)aw1.x; pw[5] = (__bf16)aw1.y; pw[6] = (__bf16)aw1.z; pw[7] = (__bf16)aw1.w;
    pb[0] = (__bf16)ab0.x; pb[1] = (__bf16)ab0.y; pb[2] = (__bf16)ab0.z; pb[3] = (__bf16)ab0.w;
    pb[4] = (__bf16)ab1.x; pb[5] = (__bf16)ab1.y; pb[6] = (__bf16)ab1.z; pb[7] = (__bf16)ab1.w;
    const int lp = r + lhoff;
    *(v8bf*)&xw[((kcw)     * 32 + lp) * 16 + ebase] = pw;
    *(v8bf*)&xw[((8 + kcw) * 32 + lp) * 16 + ebase] = pb;
  }
  __syncthreads();

  // ---- Phase 2: layer 1, [16x512] x [512x32] via bf16 WMMA -------------
  v8f acc0 = {}, acc1 = {};
  #pragma unroll 4
  for (int kc = 0; kc < KC1; ++kc) {
    v16bf a  = *(const v16bf*)&xw[(kc * 32 + lane) * 16];
    v16bf f0 = *(const v16bf*)&w1f[((kc * 2 + 0) * 32 + lane) * 16];
    v16bf f1 = *(const v16bf*)&w1f[((kc * 2 + 1) * 32 + lane) * 16];
    acc0 = __builtin_amdgcn_wmma_f32_16x16x32_bf16(false, a, false, f0,
                                                   (short)0, acc0, false, false);
    acc1 = __builtin_amdgcn_wmma_f32_16x16x32_bf16(false, a, false, f1,
                                                   (short)0, acc1, false, false);
  }
  // C/D layout: lane holds col n0 = lane&15, VGPR v holds row v + 8*(lane>>4).
  const int n0 = lane & 15;
  const int mb = (lane >> 4) * 8;
  {
    const float bias0 = b1[n0];
    const float bias1 = b1[16 + n0];
    float* hp = &hs1[wave][0];
    #pragma unroll
    for (int v = 0; v < 8; ++v) {
      hp[(mb + v) * 32 + n0]      = fmaxf(acc0[v] + bias0, 0.f);
      hp[(mb + v) * 32 + 16 + n0] = fmaxf(acc1[v] + bias1, 0.f);
    }
  }
  __syncthreads();

  // ---- Phase 3: layer 2, [16x32] x [32x32] = 2 WMMAs (K=32 exact) ------
  {
    const int m  = lane & 15;
    const int kb = (lane >> 4) * 8;
    const float* hr = &hs1[wave][m * 32];
    v16bf a2;
    #pragma unroll
    for (int j = 0; j < 8; ++j) {
      a2[j]     = (__bf16)hr[kb + j];
      a2[8 + j] = (__bf16)hr[16 + kb + j];
    }
    v16bf g0 = *(const v16bf*)&w2f[(0 * 32 + lane) * 16];
    v16bf g1 = *(const v16bf*)&w2f[(1 * 32 + lane) * 16];
    v8f d0 = {}, d1 = {};
    d0 = __builtin_amdgcn_wmma_f32_16x16x32_bf16(false, a2, false, g0,
                                                 (short)0, d0, false, false);
    d1 = __builtin_amdgcn_wmma_f32_16x16x32_bf16(false, a2, false, g1,
                                                 (short)0, d1, false, false);
    const float bias0 = b2[n0];
    const float bias1 = b2[16 + n0];
    float* hp2 = &hs2[wave][0];
    #pragma unroll
    for (int v = 0; v < 8; ++v) {
      hp2[(mb + v) * 32 + n0]      = fmaxf(d0[v] + bias0, 0.f);
      hp2[(mb + v) * 32 + 16 + n0] = fmaxf(d1[v] + bias1, 0.f);
    }
  }
  __syncthreads();

  // ---- Phase 4: layer 3, dot-32 per row --------------------------------
  if (lane < 16) {
    const float* hr2 = &hs2[wave][lane * 32];
    float acc = b3[0];
    #pragma unroll
    for (int n = 0; n < 32; ++n) acc += hr2[n] * w3[n];
    out[row_base + lane] = acc;
  }
}

extern "C" void kernel_launch(void* const* d_in, const int* in_sizes, int n_in,
                              void* d_out, int out_size, void* d_ws, size_t ws_size,
                              hipStream_t stream) {
  const int* widx = (const int*)d_in[0];
  const int* bidx = (const int*)d_in[1];
  const float* wtab = (const float*)d_in[2];
  const float* btab = (const float*)d_in[3];
  const float* w1 = (const float*)d_in[4];
  const float* b1 = (const float*)d_in[5];
  const float* w2 = (const float*)d_in[6];
  const float* b2 = (const float*)d_in[7];
  const float* w3 = (const float*)d_in[8];
  const float* b3 = (const float*)d_in[9];
  float* out = (float*)d_out;

  const int B = in_sizes[0] / NF;         // 32768
  const int rows_per_block = WPB * RPW;   // 64
  dim3 grid((B + rows_per_block - 1) / rows_per_block);
  dim3 block(WPB * 32);
  nnue_fused_kernel<<<grid, block, 0, stream>>>(
      widx, bidx, wtab, btab, w1, b1, w2, b2, w3, b3, out);
}
